// GIN_3350074491205
// MI455X (gfx1250) — matrix operand order repaired and checked by reading, not compiled, not run on hardware
//
#include <hip/hip_runtime.h>

// ---------------------------------------------------------------------------
// Types for CDNA5 WMMA (wave32). v16bf = 16 bf16 = 8 VGPRs per lane.
// ---------------------------------------------------------------------------
typedef __attribute__((ext_vector_type(16))) __bf16   v16bf;
typedef __attribute__((ext_vector_type(8)))  float    v8f;
typedef __attribute__((ext_vector_type(4)))  unsigned u32x4;

union FragBf { v16bf v; u32x4 q[2]; };   // trivial members -> legal union

#define D 128
#define NGRAPHS 500
#define BN_EPS 1e-5f
#define MTILE 64   // rows per workgroup (4 x 16-row WMMA subtiles)

// ---------------------------------------------------------------------------
// Pack W1/W2 (fp32, row-major [l][k][n]) into bf16 WMMA B-fragment order:
//   packed[((nt*4 + kb)*32 + lane)*16 + e]  where
//   n = nt*16 + lane%16,  k = kb*32 + (lane>=16 ? 8 : 0) + (e&7) + (e>=8 ? 16 : 0)
// 6 matrices: m = l*2 + {0:W1, 1:W2}
// ---------------------------------------------------------------------------
__global__ void pack_weights(const float* __restrict__ W1,
                             const float* __restrict__ W2,
                             __bf16* __restrict__ packed) {
  int idx = blockIdx.x * blockDim.x + threadIdx.x;
  if (idx >= 6 * D * D) return;
  int m    = idx / (D * D);
  int pos  = idx % (D * D);
  int e    = pos & 15;
  int lane = (pos >> 4) & 31;
  int kb   = (pos >> 9) & 3;
  int nt   = pos >> 11;
  int n = nt * 16 + (lane & 15);
  int k = kb * 32 + ((lane >> 4) ? 8 : 0) + (e & 7) + ((e >> 3) ? 16 : 0);
  int l = m >> 1;
  const float* W = (m & 1) ? W2 : W1;
  packed[idx] = (__bf16)W[(l * D + k) * D + n];
}

// ---------------------------------------------------------------------------
// GIN aggregation: agg[dst] += h[src].  32 lanes per edge, float4 per lane.
// ---------------------------------------------------------------------------
__global__ void edge_scatter(const float* __restrict__ h,
                             const int* __restrict__ src,
                             const int* __restrict__ dst,
                             float* __restrict__ agg, int nEdges) {
  int idx = blockIdx.x * blockDim.x + threadIdx.x;
  int e = idx >> 5;
  if (e >= nEdges) return;
  int f = (idx & 31) << 2;
  int s = src[e], d = dst[e];
  const float4 v = *(const float4*)(h + (size_t)s * D + f);
  float* a = agg + (size_t)d * D + f;
  atomicAdd(a + 0, v.x);
  atomicAdd(a + 1, v.y);
  atomicAdd(a + 2, v.z);
  atomicAdd(a + 3, v.w);
}

// h_in = h + agg, cast to bf16 (WMMA A operand)
__global__ void build_hin(const float* __restrict__ h,
                          const float* __restrict__ agg,
                          __bf16* __restrict__ hin, int n) {
  int i = blockIdx.x * blockDim.x + threadIdx.x;
  if (i < n) hin[i] = (__bf16)(h[i] + agg[i]);
}

// ---------------------------------------------------------------------------
// Fused MLP:  h2 = relu( relu(hin @ W1 + b1) @ W2 + b2 )
// One block = 64 rows; 8 waves each own one 16-col tile and 4 M-subtiles.
// B fragments loaded once per kb and reused across the 4 subtiles.
// h1 staged in LDS as bf16 (never hits HBM).
// BN statistics (sum / sumsq per feature) fused into the epilogue.
// ---------------------------------------------------------------------------
__global__ void __launch_bounds__(256)
gin_mlp_wmma(const __bf16* __restrict__ hin,
             const __bf16* __restrict__ w1p,
             const __bf16* __restrict__ w2p,
             const float* __restrict__ b1,
             const float* __restrict__ b2,
             float* __restrict__ h2,
             float* __restrict__ sums, int nNodes) {
  __shared__ __bf16 lds_h1[MTILE * D];     // 16 KB

  const int wave = threadIdx.x >> 5;       // 0..7 -> column tile
  const int lane = threadIdx.x & 31;
  const int col  = lane & 15;
  const int half = lane >> 4;              // 0: lanes 0-15, 1: lanes 16-31
  const int n    = wave * 16 + col;        // output column (feature)
  const int m0   = blockIdx.x * MTILE;
  const int mbase = half * 8;              // D-matrix row group per lane half

  v8f acc[4];

  // ---- GEMM1: h1 = relu(hin @ W1 + b1) ----
  {
    float bias = b1[n];
#pragma unroll
    for (int t = 0; t < 4; ++t)
#pragma unroll
      for (int r = 0; r < 8; ++r) acc[t][r] = bias;
  }
#pragma unroll
  for (int kb = 0; kb < 4; ++kb) {
    FragBf b;
    b.v = *(const v16bf*)(w1p + (((wave * 4 + kb) * 32 + lane) << 4));
#pragma unroll
    for (int t = 0; t < 4; ++t) {
      int arow = m0 + t * 16 + col;
      if (arow >= nNodes) arow = nNodes - 1;  // uniform clamp (EXEC all-1)
      FragBf a;
      const __bf16* ap = hin + (size_t)arow * D + kb * 32 + half * 8;
      a.q[0] = *(const u32x4*)(ap);           // K = base .. base+7
      a.q[1] = *(const u32x4*)(ap + 16);      // K = base+16 .. base+23
      acc[t] = __builtin_amdgcn_wmma_f32_16x16x32_bf16(
                   false, a.v, false, b.v, (short)0, acc[t], false, false);
    }
  }
#pragma unroll
  for (int t = 0; t < 4; ++t)
#pragma unroll
    for (int r = 0; r < 8; ++r)
      lds_h1[(t * 16 + mbase + r) * D + n] = (__bf16)fmaxf(acc[t][r], 0.f);
  __syncthreads();

  // ---- GEMM2: h2 = relu(h1 @ W2 + b2) ----
  {
    float bias = b2[n];
#pragma unroll
    for (int t = 0; t < 4; ++t)
#pragma unroll
      for (int r = 0; r < 8; ++r) acc[t][r] = bias;
  }
#pragma unroll
  for (int kb = 0; kb < 4; ++kb) {
    FragBf b;
    b.v = *(const v16bf*)(w2p + (((wave * 4 + kb) * 32 + lane) << 4));
#pragma unroll
    for (int t = 0; t < 4; ++t) {
      FragBf a;
      const __bf16* ap = &lds_h1[(t * 16 + col) * D + kb * 32 + half * 8];
      a.q[0] = *(const u32x4*)(ap);           // ds_load_b128
      a.q[1] = *(const u32x4*)(ap + 16);
      acc[t] = __builtin_amdgcn_wmma_f32_16x16x32_bf16(
                   false, a.v, false, b.v, (short)0, acc[t], false, false);
    }
  }

  // ---- epilogue: outer ReLU, store h2, fused BN statistics ----
  float s = 0.f, s2 = 0.f;
#pragma unroll
  for (int t = 0; t < 4; ++t) {
#pragma unroll
    for (int r = 0; r < 8; ++r) {
      int row = m0 + t * 16 + mbase + r;
      if (row < nNodes) {
        float v = fmaxf(acc[t][r], 0.f);      // outer ReLU
        h2[(size_t)row * D + n] = v;
        s  += v;
        s2 += v * v;
      }
    }
  }
  // combine the two lane-halves that hold the same feature column
  s  += __shfl_xor(s, 16);
  s2 += __shfl_xor(s2, 16);
  if (half == 0) {                            // one atomic pair per feature/block
    atomicAdd(&sums[n], s);
    atomicAdd(&sums[D + n], s2);
  }
}

// ---------------------------------------------------------------------------
// BatchNorm apply + write xs slice of d_out + per-graph pooling (atomic)
// + produce next layer's h.
// ---------------------------------------------------------------------------
__global__ void bn_apply(const float* __restrict__ h2,
                         const float* __restrict__ sums,
                         const float* __restrict__ gamma,
                         const float* __restrict__ beta,
                         const int* __restrict__ gid,
                         float* __restrict__ hout,
                         float* __restrict__ out,
                         int layer, int nNodes) {
  int i = blockIdx.x * blockDim.x + threadIdx.x;
  if (i >= nNodes * D) return;
  int node = i >> 7, f = i & (D - 1);
  float invN = 1.f / (float)nNodes;
  float mean = sums[f] * invN;
  float var  = sums[D + f] * invN - mean * mean;
  float inv  = rsqrtf(var + BN_EPS);
  float y = gamma[f] * (h2[i] - mean) * inv + beta[f];
  hout[i] = y;
  out[(size_t)NGRAPHS * 3 * D + (size_t)node * 3 * D + layer * D + f] = y;
  atomicAdd(&out[(size_t)gid[node] * 3 * D + layer * D + f], y);
}

// ---------------------------------------------------------------------------
extern "C" void kernel_launch(void* const* d_in, const int* in_sizes, int n_in,
                              void* d_out, int out_size, void* d_ws, size_t ws_size,
                              hipStream_t stream) {
  const float* x     = (const float*)d_in[0];
  const int*   esrc  = (const int*)d_in[1];
  const int*   edst  = (const int*)d_in[2];
  const int*   gid   = (const int*)d_in[3];
  const float* W1    = (const float*)d_in[4];
  const float* b1    = (const float*)d_in[5];
  const float* W2    = (const float*)d_in[6];
  const float* b2    = (const float*)d_in[7];
  const float* gamma = (const float*)d_in[8];
  const float* beta  = (const float*)d_in[9];
  float* out = (float*)d_out;

  const int nNodes = in_sizes[0] / D;      // 50000
  const int nEdges = in_sizes[1];          // 1.6M

  // workspace carve (256B aligned)
  size_t off = 0;
  auto carve = [&](size_t bytes) -> void* {
    void* p = (char*)d_ws + off;
    off += (bytes + 255) & ~(size_t)255;
    return p;
  };
  float*  h    = (float*)carve((size_t)nNodes * D * 4);
  float*  agg  = (float*)carve((size_t)nNodes * D * 4);  // aliased as h2
  __bf16* hin  = (__bf16*)carve((size_t)nNodes * D * 2);
  __bf16* wp   = (__bf16*)carve((size_t)6 * D * D * 2);
  float*  sums = (float*)carve(2 * D * 4);

  // h = x ; zero pooled-output region of d_out
  hipMemcpyAsync(h, x, (size_t)nNodes * D * 4, hipMemcpyDeviceToDevice, stream);
  hipMemsetAsync(out, 0, (size_t)NGRAPHS * 3 * D * 4, stream);

  pack_weights<<<(6 * D * D + 255) / 256, 256, 0, stream>>>(W1, W2, wp);

  const int elems = nNodes * D;
  for (int l = 0; l < 3; ++l) {
    hipMemsetAsync(agg, 0, (size_t)nNodes * D * 4, stream);
    hipMemsetAsync(sums, 0, 2 * D * 4, stream);

    unsigned scThreads = (unsigned)nEdges * 32u;
    edge_scatter<<<(scThreads + 255) / 256, 256, 0, stream>>>(h, esrc, edst, agg, nEdges);

    build_hin<<<(elems + 255) / 256, 256, 0, stream>>>(h, agg, hin, elems);

    float* h2 = agg;                       // agg no longer needed: reuse
    gin_mlp_wmma<<<(nNodes + MTILE - 1) / MTILE, 256, 0, stream>>>(
        hin, wp + (size_t)(l * 2 + 0) * D * D, wp + (size_t)(l * 2 + 1) * D * D,
        b1 + l * D, b2 + l * D, h2, sums, nNodes);

    bn_apply<<<(elems + 255) / 256, 256, 0, stream>>>(
        h2, sums, gamma + l * D, beta + l * D, gid, h, out, l, nNodes);
  }
}
